// Q_LinearBottleneck_5909874999648
// MI455X (gfx1250) — 8-bit matrix operand order repaired and checked by reading, compiled, not run on hardware
//
#include <hip/hip_runtime.h>
#include <cstdint>
#include <cstddef>

#define BATCH 64
#define CIN   96
#define CMID  576
#define COUT  96
#define HDIM  28
#define WDIM  28
#define HW    784
#define NPIX  50176
#define KPAD  128
#define NBITF 127.0f
#define EPSF  1e-5f

typedef int v8i __attribute__((ext_vector_type(8)));

__global__ void prep_w_pointwise(const float* __restrict__ w,
                                 const float* __restrict__ g,
                                 const float* __restrict__ bb,
                                 const float* __restrict__ m,
                                 const float* __restrict__ v,
                                 signed char* __restrict__ wq,
                                 float* __restrict__ ws,
                                 float* __restrict__ bf,
                                 int nOut, int nIn, int ldq) {
  int oc = blockIdx.x * blockDim.x + threadIdx.x;
  if (oc >= nOut) return;
  float sc = g[oc] * rsqrtf(v[oc] + EPSF);
  float amax = 0.f;
  for (int i = 0; i < nIn; ++i)
    amax = fmaxf(amax, fabsf(w[(size_t)oc * nIn + i] * sc));
  float s   = amax / NBITF;
  float inv = (s > 0.f) ? (1.f / s) : 0.f;
  for (int i = 0; i < nIn; ++i) {
    int q = (int)rintf(w[(size_t)oc * nIn + i] * sc * inv);
    q = max(-127, min(127, q));
    wq[(size_t)oc * ldq + i] = (signed char)q;
  }
  for (int i = nIn; i < ldq; ++i) wq[(size_t)oc * ldq + i] = 0;
  ws[oc] = s;
  bf[oc] = bb[oc] - m[oc] * sc;
}

__global__ void prep_w_dw(const float* __restrict__ w,
                          const float* __restrict__ g,
                          const float* __restrict__ bb,
                          const float* __restrict__ m,
                          const float* __restrict__ v,
                          signed char* __restrict__ wqT,
                          float* __restrict__ ws,
                          float* __restrict__ bf) {
  int c = blockIdx.x * blockDim.x + threadIdx.x;
  if (c >= CMID) return;
  float sc = g[c] * rsqrtf(v[c] + EPSF);
  float amax = 0.f;
  for (int t = 0; t < 9; ++t) amax = fmaxf(amax, fabsf(w[c * 9 + t] * sc));
  float s   = amax / NBITF;
  float inv = (s > 0.f) ? (1.f / s) : 0.f;
  for (int t = 0; t < 9; ++t) {
    int q = (int)rintf(w[c * 9 + t] * sc * inv);
    q = max(-127, min(127, q));
    wqT[(size_t)t * CMID + c] = (signed char)q;
  }
  ws[c] = s;
  bf[c] = bb[c] - m[c] * sc;
}

__global__ void prep_scales(const float* __restrict__ r0,
                            const float* __restrict__ r1,
                            const float* __restrict__ r2,
                            const float* __restrict__ r3,
                            const int* __restrict__ cl,
                            const float* __restrict__ ws1,
                            const float* __restrict__ ws2,
                            const float* __restrict__ ws3,
                            float* __restrict__ sc1,
                            float* __restrict__ sc2,
                            float* __restrict__ sc3,
                            float* __restrict__ cst) {
  int i = blockIdx.x * blockDim.x + threadIdx.x;
  int c = *cl;
  float s0 = r0[c] / NBITF;
  float s1 = r1[c] / NBITF;
  float s2 = r2[c] / NBITF;
  float s3 = r3[c] / NBITF;
  if (i < CMID) {
    sc1[i] = s0 * ws1[i];
    sc2[i] = s1 * ws2[i];
  }
  if (i < COUT) sc3[i] = s2 * ws3[i];
  if (i == 0) {
    cst[0] = 1.f / s0;
    cst[1] = 1.f / s1;
    cst[2] = 1.f / s2;
    cst[3] = 1.f / s3;
    cst[4] = s3;
  }
}

__global__ void quant_x(const float* __restrict__ x,
                        signed char* __restrict__ xq,
                        const float* __restrict__ cst) {
  int idx = blockIdx.x * blockDim.x + threadIdx.x;
  if (idx >= NPIX * KPAD) return;
  int n = idx >> 7;
  int c = idx & (KPAD - 1);
  signed char q = 0;
  if (c < CIN) {
    float inv_s0 = cst[0];
    int b = n / HW, hw = n - b * HW;
    float val = x[((size_t)b * CIN + c) * HW + hw];
    float t = fminf(fmaxf(rintf(val * inv_s0), -NBITF), NBITF);
    q = (signed char)(int)t;
  }
  xq[idx] = q;
}

static __device__ __forceinline__ v8i load_a_frag_lds(const signed char* base,
                                                      int lda, int k0, int lane) {
  const signed char* p = base + (size_t)(lane & 15) * lda + k0 + ((lane >> 4) << 4);
  int2 c0 = *(const int2*)(p);
  int2 c1 = *(const int2*)(p + 8);
  int2 c2 = *(const int2*)(p + 32);
  int2 c3 = *(const int2*)(p + 40);
  v8i a;
  a[0] = c0.x; a[1] = c0.y; a[2] = c1.x; a[3] = c1.y;
  a[4] = c2.x; a[5] = c2.y; a[6] = c3.x; a[7] = c3.y;
  return a;
}

static __device__ __forceinline__ v8i load_b_frag(const signed char* __restrict__ base,
                                                  int ldb, int n0, int k0, int lane) {
  const signed char* p = base + (size_t)(n0 + (lane & 15)) * ldb + k0 + ((lane >> 4) << 4);
  int4 d0 = *(const int4*)(p);
  int4 d1 = *(const int4*)(p + 32);
  v8i b;
  b[0] = d0.x; b[1] = d0.y; b[2] = d0.z; b[3] = d0.w;
  b[4] = d1.x; b[5] = d1.y; b[6] = d1.z; b[7] = d1.w;
  return b;
}

__global__ void __launch_bounds__(256)
conv1_gemm(const signed char* __restrict__ w1q,
           const signed char* __restrict__ xq,
           const float* __restrict__ sc1,
           const float* __restrict__ bf1,
           signed char* __restrict__ a1q,
           const float* __restrict__ cst) {
  __shared__ __align__(16) signed char As[16 * KPAD];

  const int tid   = threadIdx.x;
  const int lane  = tid & 31;
  const int wave  = tid >> 5;
  const int tileM = blockIdx.y;

  {
    const int2* src = (const int2*)(w1q + (size_t)tileM * 16 * KPAD);
    ((int2*)As)[tid] = src[tid];
  }
  __syncthreads();

  const int tileN0 = (blockIdx.x * 8 + wave) * 4;

  v8i acc[4];
#pragma unroll
  for (int t = 0; t < 4; ++t) acc[t] = (v8i){0, 0, 0, 0, 0, 0, 0, 0};

#pragma unroll
  for (int ks = 0; ks < 2; ++ks) {
    v8i a = load_a_frag_lds(As, KPAD, ks * 64, lane);
#pragma unroll
    for (int t = 0; t < 4; ++t) {
      v8i b = load_b_frag(xq, KPAD, (tileN0 + t) * 16, ks * 64, lane);
      acc[t] = __builtin_amdgcn_wmma_i32_16x16x64_iu8(true, a, true, b, acc[t],
                                                      false, false);
    }
  }

  const float inv_s1 = cst[1];
  const int ocb = tileM * 16 + ((lane >> 4) << 3);
  float scl[8], bfl[8];
#pragma unroll
  for (int r = 0; r < 8; ++r) { scl[r] = sc1[ocb + r]; bfl[r] = bf1[ocb + r]; }

#pragma unroll
  for (int t = 0; t < 4; ++t) {
    const int n = (tileN0 + t) * 16 + (lane & 15);
    int2 pk;
    signed char* pc = (signed char*)&pk;
#pragma unroll
    for (int r = 0; r < 8; ++r) {
      float y = (float)acc[t][r] * scl[r] + bfl[r];
      y = fminf(fmaxf(y, 0.f), 6.f);
      float q = fminf(fmaxf(rintf(y * inv_s1), -NBITF), NBITF);
      pc[r] = (signed char)(int)q;
    }
    *(int2*)(a1q + (size_t)n * CMID + ocb) = pk;
  }
}

__global__ void dwconv3x3(const signed char* __restrict__ a1q,
                          const signed char* __restrict__ w2qT,
                          const float* __restrict__ sc2,
                          const float* __restrict__ bf2,
                          signed char* __restrict__ a2q,
                          const float* __restrict__ cst) {
  int idx = blockIdx.x * blockDim.x + threadIdx.x;
  if (idx >= NPIX * (CMID / 4)) return;
  int n  = idx / (CMID / 4);
  int c4 = (idx - n * (CMID / 4)) * 4;
  int b  = n / HW;
  int hw = n - b * HW;
  int h  = hw / WDIM;
  int w  = hw - h * WDIM;

  int acc0 = 0, acc1 = 0, acc2 = 0, acc3 = 0;
#pragma unroll
  for (int dy = -1; dy <= 1; ++dy) {
#pragma unroll
    for (int dx = -1; dx <= 1; ++dx) {
      int hh = h + dy, ww = w + dx;
      if (hh >= 0 && hh < HDIM && ww >= 0 && ww < WDIM) {
        size_t aoff = ((size_t)(b * HW + hh * WDIM + ww)) * CMID + c4;
        char4 av = *(const char4*)(a1q + aoff);
        char4 wv = *(const char4*)(w2qT + (size_t)((dy + 1) * 3 + (dx + 1)) * CMID + c4);
        acc0 += (int)av.x * (int)wv.x;
        acc1 += (int)av.y * (int)wv.y;
        acc2 += (int)av.z * (int)wv.z;
        acc3 += (int)av.w * (int)wv.w;
      }
    }
  }

  const float inv_s2 = cst[2];
  int accs[4] = {acc0, acc1, acc2, acc3};
  char4 outq;
  signed char* oq = (signed char*)&outq;
#pragma unroll
  for (int i = 0; i < 4; ++i) {
    int c = c4 + i;
    float y = (float)accs[i] * sc2[c] + bf2[c];
    y = fminf(fmaxf(y, 0.f), 6.f);
    float t = fminf(fmaxf(rintf(y * inv_s2), -NBITF), NBITF);
    oq[i] = (signed char)(int)t;
  }
  *(char4*)(a2q + (size_t)n * CMID + c4) = outq;
}

__global__ void __launch_bounds__(256)
conv3_gemm(const signed char* __restrict__ w3q,
           const signed char* __restrict__ a2q,
           const float* __restrict__ sc3,
           const float* __restrict__ bf3,
           const float* __restrict__ x,
           float* __restrict__ out,
           const float* __restrict__ cst) {
  __shared__ __align__(16) signed char As[16 * CMID];

  const int tid   = threadIdx.x;
  const int lane  = tid & 31;
  const int wave  = tid >> 5;
  const int tileM = blockIdx.y;

  {
    const int4* src = (const int4*)(w3q + (size_t)tileM * 16 * CMID);
    int4* dst = (int4*)As;
    for (int i = tid; i < (16 * CMID) / 16; i += 256) dst[i] = src[i];
  }
  __syncthreads();

  const int tileN0 = (blockIdx.x * 8 + wave) * 4;

  v8i acc[4];
#pragma unroll
  for (int t = 0; t < 4; ++t) acc[t] = (v8i){0, 0, 0, 0, 0, 0, 0, 0};

#pragma unroll
  for (int ks = 0; ks < 9; ++ks) {
    v8i a = load_a_frag_lds(As, CMID, ks * 64, lane);
#pragma unroll
    for (int t = 0; t < 4; ++t) {
      const signed char* bp =
          a2q + (size_t)((tileN0 + t) * 16 + (lane & 15)) * CMID + ks * 64 +
          ((lane >> 4) << 4);
      int4 d0 = *(const int4*)(bp);
      int4 d1 = *(const int4*)(bp + 32);
      if (ks < 8) __builtin_prefetch(bp + 64, 0, 3);
      v8i b;
      b[0] = d0.x; b[1] = d0.y; b[2] = d0.z; b[3] = d0.w;
      b[4] = d1.x; b[5] = d1.y; b[6] = d1.z; b[7] = d1.w;
      acc[t] = __builtin_amdgcn_wmma_i32_16x16x64_iu8(true, a, true, b, acc[t],
                                                      false, false);
    }
  }

  const float inv_s3 = cst[3];
  const float s3 = cst[4];
  const int ocb = tileM * 16 + ((lane >> 4) << 3);
  float scl[8], bfl[8];
#pragma unroll
  for (int r = 0; r < 8; ++r) { scl[r] = sc3[ocb + r]; bfl[r] = bf3[ocb + r]; }

#pragma unroll
  for (int t = 0; t < 4; ++t) {
    const int n  = (tileN0 + t) * 16 + (lane & 15);
    const int b  = n / HW;
    const int hw = n - b * HW;
    const float* xp = x + ((size_t)b * COUT + ocb) * HW + hw;
    float* op = out + ((size_t)b * COUT + ocb) * HW + hw;
#pragma unroll
    for (int r = 0; r < 8; ++r) {
      float y = (float)acc[t][r] * scl[r] + bfl[r] + xp[(size_t)r * HW];
      float q = fminf(fmaxf(rintf(y * inv_s3), -NBITF), NBITF) * s3;
      op[(size_t)r * HW] = q;
    }
  }
}

__global__ void write_s3(float* __restrict__ out_tail,
                         const float* __restrict__ cst) {
  if (threadIdx.x == 0 && blockIdx.x == 0) out_tail[0] = cst[4];
}

extern "C" void kernel_launch(void* const* d_in, const int* in_sizes, int n_in,
                              void* d_out, int out_size, void* d_ws, size_t ws_size,
                              hipStream_t stream) {
  const float* x  = (const float*)d_in[0];
  const float* w1 = (const float*)d_in[1];
  const float* g1 = (const float*)d_in[2];
  const float* b1 = (const float*)d_in[3];
  const float* m1 = (const float*)d_in[4];
  const float* v1 = (const float*)d_in[5];
  const float* w2 = (const float*)d_in[6];
  const float* g2 = (const float*)d_in[7];
  const float* b2 = (const float*)d_in[8];
  const float* m2 = (const float*)d_in[9];
  const float* v2 = (const float*)d_in[10];
  const float* w3 = (const float*)d_in[11];
  const float* g3 = (const float*)d_in[12];
  const float* b3 = (const float*)d_in[13];
  const float* m3 = (const float*)d_in[14];
  const float* v3 = (const float*)d_in[15];
  const float* r0 = (const float*)d_in[16];
  const float* r1 = (const float*)d_in[17];
  const float* r2 = (const float*)d_in[18];
  const float* r3 = (const float*)d_in[19];
  const int*   cl = (const int*)d_in[20];
  float* out = (float*)d_out;

  uint8_t* wsp = (uint8_t*)d_ws;
  auto carve = [&](size_t sz) -> void* {
    void* p = (void*)wsp;
    wsp += (sz + 255) & ~(size_t)255;
    return p;
  };
  signed char* w1q  = (signed char*)carve((size_t)CMID * KPAD);
  float*       ws1  = (float*)carve(CMID * sizeof(float));
  float*       bf1  = (float*)carve(CMID * sizeof(float));
  signed char* w2qT = (signed char*)carve((size_t)9 * CMID);
  float*       ws2  = (float*)carve(CMID * sizeof(float));
  float*       bf2  = (float*)carve(CMID * sizeof(float));
  signed char* w3q  = (signed char*)carve((size_t)COUT * CMID);
  float*       ws3  = (float*)carve(COUT * sizeof(float));
  float*       bf3  = (float*)carve(COUT * sizeof(float));
  float*       sc1  = (float*)carve(CMID * sizeof(float));
  float*       sc2  = (float*)carve(CMID * sizeof(float));
  float*       sc3  = (float*)carve(COUT * sizeof(float));
  float*       cst  = (float*)carve(8 * sizeof(float));
  signed char* xq   = (signed char*)carve((size_t)NPIX * KPAD);
  signed char* a1q  = (signed char*)carve((size_t)NPIX * CMID);
  signed char* a2q  = (signed char*)carve((size_t)NPIX * CMID);
  (void)ws_size; (void)in_sizes; (void)n_in; (void)out_size;

  prep_w_pointwise<<<(CMID + 63) / 64, 64, 0, stream>>>(w1, g1, b1, m1, v1,
                                                        w1q, ws1, bf1, CMID, CIN, KPAD);
  prep_w_dw<<<(CMID + 63) / 64, 64, 0, stream>>>(w2, g2, b2, m2, v2, w2qT, ws2, bf2);
  prep_w_pointwise<<<(COUT + 63) / 64, 64, 0, stream>>>(w3, g3, b3, m3, v3,
                                                        w3q, ws3, bf3, COUT, CMID, CMID);

  prep_scales<<<(CMID + 255) / 256, 256, 0, stream>>>(r0, r1, r2, r3, cl,
                                                      ws1, ws2, ws3,
                                                      sc1, sc2, sc3, cst);

  {
    int total = NPIX * KPAD;
    quant_x<<<(total + 255) / 256, 256, 0, stream>>>(x, xq, cst);
  }

  conv1_gemm<<<dim3(NPIX / 512, CMID / 16), 256, 0, stream>>>(
      w1q, xq, sc1, bf1, a1q, cst);

  {
    int total = NPIX * (CMID / 4);
    dwconv3x3<<<(total + 255) / 256, 256, 0, stream>>>(
        a1q, w2qT, sc2, bf2, a2q, cst);
  }

  conv3_gemm<<<dim3(NPIX / 512, COUT / 16), 256, 0, stream>>>(
      w3q, a2q, sc3, bf3, x, out, cst);

  write_s3<<<1, 1, 0, stream>>>(out + (size_t)BATCH * COUT * HW, cst);
}
